// GNNMLP_6236292513987
// MI455X (gfx1250) — matrix operand order repaired and checked by reading, not compiled, not run on hardware
//
#include <hip/hip_runtime.h>

// ---------------------------------------------------------------------------
// GNN forward for gfx1250 (MI455X). All dense GEMMs on the WMMA path
// (v_wmma_f32_16x16x32_f16, f32 accumulate). Operands are zero-padded at
// build time so the GEMM hot loop has no bounds checks: vectorized 16B
// global loads -> ds_store_b128 -> ds_load_b128 fragments -> 2x WMMA.
// ---------------------------------------------------------------------------

typedef __attribute__((ext_vector_type(16))) _Float16 v16h;
typedef __attribute__((ext_vector_type(8)))  float    v8f;

#define BN_EPS 1e-5f

enum { ACT_NONE = 0, ACT_LRELU = 1, ACT_RELU = 2 };

// order-preserving float <-> uint for atomic max on signed floats
__device__ __forceinline__ unsigned f32_enc(float f) {
    unsigned b = __float_as_uint(f);
    return (b & 0x80000000u) ? ~b : (b | 0x80000000u);
}
__device__ __forceinline__ float f32_dec(unsigned e) {
    return (e & 0x80000000u) ? __uint_as_float(e ^ 0x80000000u)
                             : __uint_as_float(~e);
}

// ---------------------------------------------------------------------------
// WMMA GEMM: C[M,Nc] = act(A[M,Kp](f16, zero-padded) @ W + bias)
//   A   : row-major, lda = Kp (multiple of 32), rows padded to multiple of 64
//   Bt  : weights pre-transposed+padded: Bt[n][k], n < Npad(>= col coverage),
//         ldb = Kp, zero outside the true K x Nc extent
// Block tile 64x64, 8 waves (4 row x 2 col), each wave: 16x32 C = 2 WMMAs.
// Epilogue modes: f32 out, f16 out (writes zeros into its column padding),
// and fused attention scatter: scatterOut[dst[row]][col] += ex/sum * v.
// ---------------------------------------------------------------------------
__global__ __launch_bounds__(256)
void k_gemm_wmma(const _Float16* __restrict__ A,
                 const _Float16* __restrict__ Bt,
                 const float* __restrict__ bias,
                 float* __restrict__ outF, int ldoF,
                 _Float16* __restrict__ outH, int ldoH,
                 int M, int Ncols, int Kp, int act,
                 const int* __restrict__ dstIdx,
                 const float* __restrict__ exBuf,
                 const float* __restrict__ sumBuf,
                 float* __restrict__ scatterOut, int ldoS)
{
    __shared__ _Float16 Ah[64][40];   // 80B row stride (multiple of 16B)
    __shared__ _Float16 Bh[64][40];   // Bh[n][k]

    const int tid  = threadIdx.x;
    const int lane = tid & 31;
    const int w    = tid >> 5;
    const int wm   = w & 3;                  // wave row tile (16 rows)
    const int wn   = w >> 2;                 // wave col tile (32 cols)
    const long row0 = (long)blockIdx.x * 64;
    const long col0 = (long)blockIdx.y * 64;

    const int g   = lane >> 4;
    const int l16 = lane & 15;

    // staging assignment: 8 halves (16B) per thread for each tile
    const int sr = tid >> 2;                 // 0..63 (A row / Bt n-row)
    const int sk = (tid & 3) * 8;            // 0,8,16,24

    const _Float16* aSrc = A  + (row0 + sr) * (long)Kp + sk;
    const _Float16* bSrc = Bt + (col0 + sr) * (long)Kp + sk;

    v8f acc0 = {}, acc1 = {};

    for (int kb = 0; kb < Kp; kb += 32) {
        // unconditional vectorized staging (operands are zero-padded)
        *(float4*)&Ah[sr][sk] = *(const float4*)(aSrc + kb);
        *(float4*)&Bh[sr][sk] = *(const float4*)(bSrc + kb);
        if (kb + 32 < Kp) {
            __builtin_prefetch(aSrc + kb + 32, 0, 1);
            __builtin_prefetch(bSrc + kb + 32, 0, 1);
        }
        __syncthreads();

        // fragments per ISA 7.12.2 wave32 layouts; pairs -> 32-bit LDS loads
        union { v16h v; unsigned u[8]; } af, bf0, bf1;
        const int arow  = wm * 16 + l16;
        const int bcol0 = wn * 32 + l16;
        #pragma unroll
        for (int j = 0; j < 8; ++j) {
            const int ka = ((j < 4) ? (2 * j) : (16 + 2 * (j - 4))) + 8 * g;
            af.u[j]  = *(const unsigned*)&Ah[arow][ka];
            const int kbv = ((j < 4) ? (2 * j) : (16 + 2 * (j - 4))) + 8 * g;
            bf0.u[j] = *(const unsigned*)&Bh[bcol0][kbv];
            bf1.u[j] = *(const unsigned*)&Bh[bcol0 + 16][kbv];
        }
        acc0 = __builtin_amdgcn_wmma_f32_16x16x32_f16(false, af.v, false, bf0.v,
                                                      (short)0, acc0, false, false);
        acc1 = __builtin_amdgcn_wmma_f32_16x16x32_f16(false, af.v, false, bf1.v,
                                                      (short)0, acc1, false, false);
        __syncthreads();
    }

    // epilogue: C lane j -> (M = j + 8g, N = lane&15)
    #pragma unroll
    for (int t = 0; t < 2; ++t) {
        const long col = col0 + wn * 32 + t * 16 + l16;
        #pragma unroll
        for (int j = 0; j < 8; ++j) {
            const long row = row0 + wm * 16 + j + 8 * g;
            if (row >= M) continue;
            float v = t ? acc1[j] : acc0[j];
            if (col < Ncols) {
                if (bias) v += bias[col];
                if (act == ACT_LRELU)     v = v > 0.f ? v : 0.01f * v;
                else if (act == ACT_RELU) v = v > 0.f ? v : 0.f;
            }
            if (dstIdx) {
                if (col < Ncols) {
                    const int d = dstIdx[row];
                    const float a = exBuf[row] / sumBuf[d];
                    atomicAdd(&scatterOut[(long)d * ldoS + col], a * v);
                }
            } else {
                if (outF && col < Ncols) outF[row * (long)ldoF + col] = v;
                if (outH && col < ldoH)                       // zero the K-padding
                    outH[row * (long)ldoH + col] = (col < Ncols) ? (_Float16)v
                                                                 : (_Float16)0.0f;
            }
        }
    }
}

// ---------------------------------------------------------------------------
// builders / elementwise
// ---------------------------------------------------------------------------
__global__ void k_cvt_f16(const float* __restrict__ in, _Float16* __restrict__ out, long n) {
    long i = (long)blockIdx.x * blockDim.x + threadIdx.x;
    if (i < n) out[i] = (_Float16)in[i];
}

// weights: W[K,Nc] f32 -> Bt[Npad][Kp] f16 transposed, zero-padded
__global__ void k_cvt_w_t(const float* __restrict__ W, _Float16* __restrict__ out,
                          int K, int Nc, int Kp, int Npad) {
    long idx = (long)blockIdx.x * blockDim.x + threadIdx.x;
    if (idx >= (long)Npad * Kp) return;
    const int n = (int)(idx / Kp);
    const int k = (int)(idx % Kp);
    out[idx] = (n < Nc && k < K) ? (_Float16)W[(long)k * Nc + n] : (_Float16)0.0f;
}

__global__ void k_fill_f32(float* p, long n, float v) {
    long i = (long)blockIdx.x * blockDim.x + threadIdx.x;
    if (i < n) p[i] = v;
}

__global__ void k_fill_u32(unsigned* p, long n, unsigned v) {
    long i = (long)blockIdx.x * blockDim.x + threadIdx.x;
    if (i < n) p[i] = v;
}

// A_e1[e,0:Dn]=node_feats[src[e]], A_e1[e,Dn:Dn+De]=edge_feats[e]  (f16)
__global__ void k_build_Ae1(const float* __restrict__ nf, const float* __restrict__ ef,
                            const int* __restrict__ src, _Float16* __restrict__ out,
                            long E, int Dn, int De) {
    const int Kt = Dn + De;
    long idx = (long)blockIdx.x * blockDim.x + threadIdx.x;
    if (idx >= E * (long)Kt) return;
    const long e = idx / Kt;
    const int  k = (int)(idx % Kt);
    float v = (k < Dn) ? nf[(long)src[e] * Dn + k]
                       : ef[e * (long)De + (k - Dn)];
    out[idx] = (_Float16)v;
}

// wave-per-edge: logits[e] = lrelu(hv[dst].W[0:G] + he1[e].W[G:2G] + b)
__global__ __launch_bounds__(256)
void k_logits(const float* __restrict__ hv, const _Float16* __restrict__ he1, int ldh,
              const float* __restrict__ W, const float* __restrict__ b,
              const int* __restrict__ dst, float* __restrict__ out, long E, int G) {
    const int lane = threadIdx.x & 31;
    const long e = (long)blockIdx.x * 8 + (threadIdx.x >> 5);
    if (e >= E) return;
    const long d = dst[e];
    float acc = 0.f;
    for (int k = lane; k < G; k += 32)
        acc += hv[d * G + k] * W[k] + (float)he1[e * ldh + k] * W[G + k];
    for (int off = 16; off; off >>= 1) acc += __shfl_xor(acc, off, 32);
    if (lane == 0) { float v = acc + b[0]; out[e] = v > 0.f ? v : 0.01f * v; }
}

// wave-per-edge: lg[e] = lrelu(h0[dst].W[0:G] + h0[src].W[G:2G] + b)
__global__ __launch_bounds__(256)
void k_lg(const float* __restrict__ h0, const float* __restrict__ W,
          const float* __restrict__ b, const int* __restrict__ src,
          const int* __restrict__ dst, float* __restrict__ out, long E, int G) {
    const int lane = threadIdx.x & 31;
    const long e = (long)blockIdx.x * 8 + (threadIdx.x >> 5);
    if (e >= E) return;
    const long d = dst[e], s = src[e];
    float acc = 0.f;
    for (int k = lane; k < G; k += 32)
        acc += h0[d * G + k] * W[k] + h0[s * G + k] * W[G + k];
    for (int off = 16; off; off >>= 1) acc += __shfl_xor(acc, off, 32);
    if (lane == 0) { float v = acc + b[0]; out[e] = v > 0.f ? v : 0.01f * v; }
}

__global__ void k_segmax(const float* __restrict__ lg, const int* __restrict__ dst,
                         unsigned* __restrict__ mx, long E) {
    long e = (long)blockIdx.x * blockDim.x + threadIdx.x;
    if (e >= E) return;
    atomicMax(&mx[dst[e]], f32_enc(lg[e]));
}

// in-place: ex[e] = exp(lg[e] - max[dst[e]]); sum[dst[e]] += ex[e]
__global__ void k_segexpsum(float* __restrict__ lg_ex, const int* __restrict__ dst,
                            const unsigned* __restrict__ mx, float* __restrict__ sum,
                            long E) {
    long e = (long)blockIdx.x * blockDim.x + threadIdx.x;
    if (e >= E) return;
    const int d = dst[e];
    float ex = __expf(lg_ex[e] - f32_dec(mx[d]));
    lg_ex[e] = ex;
    atomicAdd(&sum[d], ex);
}

// c2[dst[e], j] += (ex[e]/sum[dst[e]]) * hvp[src[e], j]
__global__ void k_scatter_msg(const float* __restrict__ hvp, const int* __restrict__ src,
                              const int* __restrict__ dst, const float* __restrict__ ex,
                              const float* __restrict__ sum, float* __restrict__ c2,
                              long E, int G) {
    long idx = (long)blockIdx.x * blockDim.x + threadIdx.x;
    if (idx >= E * (long)G) return;
    const long e = idx / G;
    const int  j = (int)(idx % G);
    const int  d = dst[e];
    const float a = ex[e] / sum[d];
    atomicAdd(&c2[(long)d * G + j], a * hvp[(long)src[e] * G + j]);
}

// out[i,0:G]=elu(c[i]); out[i,G:2G]=right[i]; out[i,2G:K2p]=0   (f16, padded)
__global__ void k_elu_concat(const float* __restrict__ c, const float* __restrict__ right,
                             _Float16* __restrict__ out, long N, int G, int K2p) {
    long idx = (long)blockIdx.x * blockDim.x + threadIdx.x;
    if (idx >= N * (long)K2p) return;
    const long i = idx / K2p;
    const int  j = (int)(idx % K2p);
    float v = 0.f;
    if (j < G)          { v = c[i * G + j]; v = v > 0.f ? v : (__expf(v) - 1.f); }
    else if (j < 2 * G) { v = right[i * G + (j - G)]; }
    out[idx] = (_Float16)v;
}

// per-column batch stats (biased variance), one block per column
__global__ __launch_bounds__(256)
void k_bn_stats(const float* __restrict__ hh, float* __restrict__ mean,
                float* __restrict__ var, int N, int G) {
    __shared__ float s1[256], s2[256];
    const int j = blockIdx.x;
    float a = 0.f, b = 0.f;
    for (int i = threadIdx.x; i < N; i += 256) {
        float v = hh[(long)i * G + j];
        a += v; b += v * v;
    }
    s1[threadIdx.x] = a; s2[threadIdx.x] = b;
    __syncthreads();
    for (int st = 128; st > 0; st >>= 1) {
        if (threadIdx.x < st) { s1[threadIdx.x] += s1[threadIdx.x + st];
                                s2[threadIdx.x] += s2[threadIdx.x + st]; }
        __syncthreads();
    }
    if (threadIdx.x == 0) {
        float m = s1[0] / N;
        mean[j] = m;
        var[j]  = s2[0] / N - m * m;
    }
}

// out = h0 + (hh - mean)*rsqrt(var+eps)*gamma + beta
__global__ void k_bn_out(const float* __restrict__ h0, const float* __restrict__ hh,
                         const float* __restrict__ mean, const float* __restrict__ var,
                         const float* __restrict__ gamma, const float* __restrict__ beta,
                         float* __restrict__ out, long N, int G) {
    long idx = (long)blockIdx.x * blockDim.x + threadIdx.x;
    if (idx >= N * (long)G) return;
    const int j = (int)(idx % G);
    float v = (hh[idx] - mean[j]) * rsqrtf(var[j] + BN_EPS) * gamma[j] + beta[j];
    out[idx] = h0[idx] + v;
}

// ---------------------------------------------------------------------------
// host side
// ---------------------------------------------------------------------------
static inline unsigned gblk(long n) { return (unsigned)((n + 255) / 256); }
static inline int rup(int v, int m) { return (v + m - 1) / m * m; }

static void launch_gemm(hipStream_t s,
                        const _Float16* A, const _Float16* Bt, const float* bias,
                        float* outF, int ldoF, _Float16* outH, int ldoH,
                        int M, int Nc, int Kp, int act,
                        const int* dstIdx = nullptr, const float* ex = nullptr,
                        const float* sum = nullptr, float* scOut = nullptr, int ldoS = 0) {
    dim3 grid((M + 63) / 64, (Nc + 63) / 64);
    k_gemm_wmma<<<grid, 256, 0, s>>>(A, Bt, bias, outF, ldoF, outH, ldoH,
                                     M, Nc, Kp, act, dstIdx, ex, sum, scOut, ldoS);
}

extern "C" void kernel_launch(void* const* d_in, const int* in_sizes, int n_in,
                              void* d_out, int out_size, void* d_ws, size_t ws_size,
                              hipStream_t stream) {
    const float* node_feats = (const float*)d_in[0];
    const float* edge_feats = (const float*)d_in[1];
    const int*   src        = (const int*)d_in[2];
    const int*   dst        = (const int*)d_in[3];
    const float* Wn  = (const float*)d_in[4];   const float* bn  = (const float*)d_in[5];
    const float* We1 = (const float*)d_in[6];   const float* be1 = (const float*)d_in[7];
    const float* We2 = (const float*)d_in[8];   const float* be2 = (const float*)d_in[9];
    const float* Wt  = (const float*)d_in[10];  const float* bt  = (const float*)d_in[11];
    const float* W1  = (const float*)d_in[12];  const float* b1  = (const float*)d_in[13];
    const float* W2  = (const float*)d_in[14];  const float* b2  = (const float*)d_in[15];
    const float* Wpe = (const float*)d_in[16];  const float* bpe = (const float*)d_in[17];
    const float* Wp  = (const float*)d_in[18];  const float* bp  = (const float*)d_in[19];
    const float* Wm1 = (const float*)d_in[20];  const float* bm1 = (const float*)d_in[21];
    const float* Wm2 = (const float*)d_in[22];  const float* bm2 = (const float*)d_in[23];
    const float* gamma = (const float*)d_in[24];
    const float* beta  = (const float*)d_in[25];
    float* out = (float*)d_out;

    const int  G   = in_sizes[5];               // 200
    const int  Dn  = in_sizes[4] / G;           // 128
    const long E   = in_sizes[2];               // 320000  (multiple of 64)
    const long N   = (long)in_sizes[0] / Dn;    // 10000
    const int  De  = (int)(in_sizes[1] / E);    // 64
    const int  K1  = Dn + De;                   // 192 (mult of 32)
    const int  Gp  = rup(G, 32);                // 224
    const int  K2p = rup(2 * G, 32);            // 416
    const int  Np  = rup(G, 64) ? rup(rup(G, 64), 64) : 64;  // weight col padding
    const int  NpW = rup(G, 64);                // 256? -> rup(200,64)=256? no: 256
    const int  Npad = rup(G, 64);               // col coverage of 64-wide tiles = 256
    (void)Np; (void)NpW;
    const long Er  = E;                         // E already mult of 64
    const long Nr  = rup((int)N, 64);           // padded row count for node matrices

    // ---- workspace (256B-aligned bump allocator) ----
    char* p = (char*)d_ws;
    auto alloc = [&](size_t bytes) -> char* {
        char* r = p; p += (bytes + 255) & ~(size_t)255; return r;
    };
    auto allocH = [&](size_t n) { return (_Float16*)alloc(sizeof(_Float16) * n); };
    auto allocF = [&](size_t n) { return (float*)alloc(sizeof(float) * n); };

    _Float16* Wn_t  = allocH((size_t)Npad * Dn);
    _Float16* We1_t = allocH((size_t)Npad * K1);
    _Float16* Wt_t  = allocH((size_t)Npad * Gp);
    _Float16* W1_t  = allocH((size_t)Npad * K2p);
    _Float16* W2_t  = allocH((size_t)Npad * Gp);
    _Float16* Wp_t  = allocH((size_t)Npad * Gp);
    _Float16* Wm1_t = allocH((size_t)Npad * K2p);
    _Float16* Wm2_t = allocH((size_t)Npad * Gp);
    _Float16* nodeA_h   = allocH((size_t)Nr * Dn);
    _Float16* Ae1_h     = allocH((size_t)Er * K1);
    _Float16* he1_h     = allocH((size_t)Er * Gp);
    float*    hv_new    = allocF((size_t)N * G);
    float*    lgbuf     = allocF((size_t)E);        // logits -> ex (in place)
    unsigned* maxbuf    = (unsigned*)alloc(sizeof(unsigned) * (size_t)N);
    float*    sumbuf    = allocF((size_t)N);
    float*    cbuf      = allocF((size_t)N * G);    // c, then c2
    _Float16* concatA_h = allocH((size_t)Nr * K2p);
    _Float16* h_h       = allocH((size_t)Nr * Gp);  // h, then hh(mid)
    float*    h0        = allocF((size_t)N * G);
    _Float16* h0_h      = allocH((size_t)Nr * Gp);
    float*    hvp       = allocF((size_t)N * G);
    float*    hh        = allocF((size_t)N * G);
    float*    mean      = allocF((size_t)G);
    float*    var       = allocF((size_t)G);
    (void)ws_size; (void)n_in; (void)out_size;

    const unsigned NEG_INF_ENC = 0x007FFFFFu;  // f32_enc(-inf)

    // ---- 0) weights -> f16, transposed + padded; node feats -> f16 ----
    k_cvt_w_t<<<gblk((long)Npad * Dn), 256, 0, stream>>>(Wn,  Wn_t,  Dn, G, Dn,  Npad);
    k_cvt_w_t<<<gblk((long)Npad * K1), 256, 0, stream>>>(We1, We1_t, K1, G, K1,  Npad);
    k_cvt_w_t<<<gblk((long)Npad * Gp), 256, 0, stream>>>(Wt,  Wt_t,  G,  G, Gp,  Npad);
    k_cvt_w_t<<<gblk((long)Npad * K2p),256, 0, stream>>>(W1,  W1_t,  2*G,G, K2p, Npad);
    k_cvt_w_t<<<gblk((long)Npad * Gp), 256, 0, stream>>>(W2,  W2_t,  G,  G, Gp,  Npad);
    k_cvt_w_t<<<gblk((long)Npad * Gp), 256, 0, stream>>>(Wp,  Wp_t,  G,  G, Gp,  Npad);
    k_cvt_w_t<<<gblk((long)Npad * K2p),256, 0, stream>>>(Wm1, Wm1_t, 2*G,G, K2p, Npad);
    k_cvt_w_t<<<gblk((long)Npad * Gp), 256, 0, stream>>>(Wm2, Wm2_t, G,  G, Gp,  Npad);
    k_cvt_f16<<<gblk(N * Dn), 256, 0, stream>>>(node_feats, nodeA_h, N * Dn);

    // ---- 1) hv_new = lrelu(node_feats @ Wn + bn)  [N,G] f32 ----
    launch_gemm(stream, nodeA_h, Wn_t, bn, hv_new, G, nullptr, 0,
                (int)N, G, Dn, ACT_LRELU);

    // ---- 2) he1 = lrelu(concat(nf[src], ef) @ We1 + be1)  [E,Gp] f16 ----
    k_build_Ae1<<<gblk(E * K1), 256, 0, stream>>>(node_feats, edge_feats, src,
                                                  Ae1_h, E, Dn, De);
    launch_gemm(stream, Ae1_h, We1_t, be1, nullptr, 0, he1_h, Gp,
                (int)E, G, K1, ACT_LRELU);

    // ---- 3) attention 1: logits, edge softmax ----
    k_logits<<<gblk(E * 32), 256, 0, stream>>>(hv_new, he1_h, Gp, We2, be2, dst,
                                               lgbuf, E, G);
    k_fill_u32<<<gblk(N), 256, 0, stream>>>(maxbuf, N, NEG_INF_ENC);
    k_fill_f32<<<gblk(N), 256, 0, stream>>>(sumbuf, N, 0.f);
    k_fill_f32<<<gblk(N * G), 256, 0, stream>>>(cbuf, N * G, 0.f);
    k_segmax<<<gblk(E), 256, 0, stream>>>(lgbuf, dst, maxbuf, E);
    k_segexpsum<<<gblk(E), 256, 0, stream>>>(lgbuf, dst, maxbuf, sumbuf, E);

    // ---- 4) c = segment_sum(a * (he1 @ Wt + bt))  -- fused scatter GEMM ----
    launch_gemm(stream, he1_h, Wt_t, bt, nullptr, 0, nullptr, 0,
                (int)E, G, Gp, ACT_NONE, dst, lgbuf, sumbuf, cbuf, G);

    // ---- 5) h0 = relu(relu(concat(elu(c), hv_new) @ W1 + b1) @ W2 + b2) ----
    k_elu_concat<<<gblk(N * K2p), 256, 0, stream>>>(cbuf, hv_new, concatA_h, N, G, K2p);
    launch_gemm(stream, concatA_h, W1_t, b1, nullptr, 0, h_h, Gp,
                (int)N, G, K2p, ACT_RELU);
    launch_gemm(stream, h_h, W2_t, b2, h0, G, h0_h, Gp,
                (int)N, G, Gp, ACT_RELU);

    // ---- 6) attention 2: lg, edge softmax (buffers reused) ----
    k_lg<<<gblk(E * 32), 256, 0, stream>>>(h0, Wpe, bpe, src, dst, lgbuf, E, G);
    k_fill_u32<<<gblk(N), 256, 0, stream>>>(maxbuf, N, NEG_INF_ENC);
    k_fill_f32<<<gblk(N), 256, 0, stream>>>(sumbuf, N, 0.f);
    k_fill_f32<<<gblk(N * G), 256, 0, stream>>>(cbuf, N * G, 0.f);
    k_segmax<<<gblk(E), 256, 0, stream>>>(lgbuf, dst, maxbuf, E);
    k_segexpsum<<<gblk(E), 256, 0, stream>>>(lgbuf, dst, maxbuf, sumbuf, E);

    // ---- 7) hvp = h0 @ Wp + bp ; c2 = segment_sum(a2 * hvp[src]) ----
    launch_gemm(stream, h0_h, Wp_t, bp, hvp, G, nullptr, 0,
                (int)N, G, Gp, ACT_NONE);
    k_scatter_msg<<<gblk(E * G), 256, 0, stream>>>(hvp, src, dst, lgbuf, sumbuf,
                                                   cbuf, E, G);

    // ---- 8) hh = relu(relu(concat(elu(c2), h0) @ Wm1 + bm1) @ Wm2 + bm2) ----
    k_elu_concat<<<gblk(N * K2p), 256, 0, stream>>>(cbuf, h0, concatA_h, N, G, K2p);
    launch_gemm(stream, concatA_h, Wm1_t, bm1, nullptr, 0, h_h, Gp,
                (int)N, G, K2p, ACT_RELU);
    launch_gemm(stream, h_h, Wm2_t, bm2, hh, G, nullptr, 0,
                (int)N, G, Gp, ACT_RELU);

    // ---- 9) batchnorm (training stats) + residual ----
    k_bn_stats<<<G, 256, 0, stream>>>(hh, mean, var, (int)N, G);
    k_bn_out<<<gblk(N * G), 256, 0, stream>>>(h0, hh, mean, var, gamma, beta,
                                              out, N, G);
}
